// CriticBatchNet_90202903151098
// MI455X (gfx1250) — matrix operand order repaired and checked by reading, not compiled, not run on hardware
//
#include <hip/hip_runtime.h>

typedef __attribute__((ext_vector_type(16))) _Float16 v16h;
typedef __attribute__((ext_vector_type(8)))  _Float16 v8h;
typedef __attribute__((ext_vector_type(8)))  float    v8f;

__device__ __forceinline__ float sigf_(float x){ return 1.0f/(1.0f+__expf(-x)); }

__device__ __forceinline__ v16h load_v16h(const _Float16* p){
  const v8h* q = (const v8h*)p;
  v8h lo = q[0], hi = q[1];
  v16h r;
#pragma unroll
  for (int i = 0; i < 8; ++i){ r[i] = lo[i]; r[i+8] = hi[i]; }
  return r;
}

__global__ void k_zero_f32(float* p, int n){
  int i = blockIdx.x*blockDim.x + threadIdx.x;
  if (i < n) p[i] = 0.f;
}

// X = relu(x @ lin0_w + lin0_b), F=3, D=32. One thread per (node, feature).
__global__ void k_lin0(const float* __restrict__ x, const float* __restrict__ w,
                       const float* __restrict__ b, float* __restrict__ X, int N){
  int t = blockIdx.x*blockDim.x + threadIdx.x;
  if (t >= N*32) return;
  int n = t >> 5, f = t & 31;
  float a = b[f];
  a += x[n*3+0]*w[0*32+f];
  a += x[n*3+1]*w[1*32+f];
  a += x[n*3+2]*w[2*32+f];
  X[t] = fmaxf(a, 0.f);
}

// hidden = relu(edge_attr @ e_w1 + e_b1), written fp16 in WMMA A-operand swizzle.
// 16-bit A 16x32 layout: lane = ((K>>3)&1)*16 + m ; half = ((K>>4)<<3) + (K&7)
__global__ void k_ehidden(const float* __restrict__ ea, const float* __restrict__ w1,
                          const float* __restrict__ b1, _Float16* __restrict__ Apack, int E){
  int t = blockIdx.x*blockDim.x + threadIdx.x;
  if (t >= E*32) return;
  int e = t >> 5, j = t & 31;
  float a = b1[j];
#pragma unroll
  for (int k = 0; k < 4; ++k) a += ea[e*4+k]*w1[k*32+j];
  a = fmaxf(a, 0.f);
  int tile = e >> 4, m = e & 15;
  int lane = ((j >> 3) & 1)*16 + m;
  int h    = ((j >> 4) << 3) + (j & 7);
  Apack[(size_t)tile*512 + lane*16 + h] = (_Float16)a;
}

// Pack e_w2 (32x1024 f32) into WMMA B-operand tiles (64 tiles of 32x16 f16).
// 16-bit B 32x16 layout: lane = ((K>>3)&1)*16 + n ; half = ((K>>4)<<3) + (K&7)
__global__ void k_packB(const float* __restrict__ w2, _Float16* __restrict__ Bpack){
  int t = blockIdx.x*blockDim.x + threadIdx.x;
  if (t >= 32*1024) return;
  int K = t >> 10, c = t & 1023;
  int tn = c >> 4, nn = c & 15;
  int lane = ((K >> 3) & 1)*16 + nn;
  int h    = ((K >> 4) << 3) + (K & 7);
  Bpack[tn*512 + lane*16 + h] = (_Float16)w2[t];
}

// W_e = hidden @ e_w2 + e_b2, via v_wmma_f32_16x16x32_f16.
// One wave per 16-edge M-tile; loops the 64 N-tiles reusing A in registers.
// Stores TRANSPOSED per edge: We[e][f][d] (f = out-feature, d = in-feature),
// so the msg kernel streams each lane's row with contiguous b128 loads.
// Product column c = gn*16 + n maps to (d = c>>5, f = c&31).
__global__ void k_we_gemm(const _Float16* __restrict__ Apack,
                          const _Float16* __restrict__ Bpack,
                          const float* __restrict__ b2,
                          _Float16* __restrict__ We, int tilesM){
  int wave = (blockIdx.x*blockDim.x + threadIdx.x) >> 5;
  int lane = threadIdx.x & 31;
  if (wave >= tilesM) return;
  v16h a = load_v16h(Apack + (size_t)wave*512 + lane*16);
  int n = lane & 15;
  int rowsel = (lane >> 4) ? 8 : 0;
  size_t ebase = (size_t)wave * 16;
#pragma unroll 4
  for (int gn = 0; gn < 64; ++gn){
    v16h b = load_v16h(Bpack + gn*512 + lane*16);
    float bias = b2[gn*16 + n];
    v8f c;
#pragma unroll
    for (int r = 0; r < 8; ++r) c[r] = bias;
    c = __builtin_amdgcn_wmma_f32_16x16x32_f16(
        /*neg_a=*/false, a, /*neg_b=*/false, b,
        /*c_mod=*/(short)0, c, /*reuse_a=*/false, /*reuse_b=*/false);
    int col = gn*16 + n;           // c index in [0,1024)
    int dtr = col >> 5;            // d
    int ftr = col & 31;            // f
    size_t toff = (size_t)ftr*32 + dtr;
#pragma unroll
    for (int r = 0; r < 8; ++r){
      size_t e = ebase + rowsel + r;
      We[e*1024 + toff] = (_Float16)c[r];
    }
  }
}

__global__ void k_deg(const int* __restrict__ dst, float* __restrict__ deg, int E){
  int e = blockIdx.x*blockDim.x + threadIdx.x;
  if (e < E) atomicAdd(&deg[dst[e]], 1.f);
}

// msg[e][f] = sum_d out[src[e]][d] * W_e[e][d][f]; scatter-add into aggr[dst[e]].
// One wave per edge; W_e stored f-major so lane f loads its 64B row with 4x b128.
__global__ void k_msg(const _Float16* __restrict__ We, const float* __restrict__ X,
                      const int* __restrict__ src, const int* __restrict__ dst,
                      float* __restrict__ aggr, int E){
  int t = blockIdx.x*blockDim.x + threadIdx.x;
  int e = t >> 5, f = t & 31;
  if (e >= E) return;
  int sn = src[e], dn = dst[e];
  float xv = X[sn*32 + f];
  const v8h* Wp = (const v8h*)(We + (size_t)e*1024 + f*32);
  v8h w0 = Wp[0], w1 = Wp[1], w2 = Wp[2], w3 = Wp[3];
  float a0 = 0.f, a1 = 0.f, a2 = 0.f, a3 = 0.f;
#pragma unroll
  for (int d = 0; d < 8; ++d){
    a0 += __shfl(xv, d,      32) * (float)w0[d];
    a1 += __shfl(xv, d + 8,  32) * (float)w1[d];
    a2 += __shfl(xv, d + 16, 32) * (float)w2[d];
    a3 += __shfl(xv, d + 24, 32) * (float)w3[d];
  }
  atomicAdd(&aggr[dn*32 + f], (a0 + a1) + (a2 + a3));
}

// m = relu(aggr/deg + out@root_w + conv_b); GRU cell update. One wave per node.
__global__ void k_gru(float* __restrict__ X, const float* __restrict__ aggr,
                      const float* __restrict__ deg,
                      const float* __restrict__ root_w, const float* __restrict__ conv_b,
                      const float* __restrict__ wih, const float* __restrict__ whh,
                      const float* __restrict__ bih, const float* __restrict__ bhh, int N){
  int t = blockIdx.x*blockDim.x + threadIdx.x;
  int n = t >> 5, f = t & 31;
  if (n >= N) return;
  float h = X[n*32 + f];
  float m = aggr[n*32 + f] / fmaxf(deg[n], 1.f) + conv_b[f];
#pragma unroll
  for (int d = 0; d < 32; ++d) m += __shfl(h, d, 32) * root_w[d*32 + f];
  m = fmaxf(m, 0.f);
  float gi_r = bih[f], gi_z = bih[32+f], gi_n = bih[64+f];
  float gh_r = bhh[f], gh_z = bhh[32+f], gh_n = bhh[64+f];
#pragma unroll
  for (int d = 0; d < 32; ++d){
    float md = __shfl(m, d, 32);
    float hd = __shfl(h, d, 32);
    const float* wi = wih + d*96;
    const float* wh = whh + d*96;
    gi_r += md*wi[f];    gi_z += md*wi[32+f];    gi_n += md*wi[64+f];
    gh_r += hd*wh[f];    gh_z += hd*wh[32+f];    gh_n += hd*wh[64+f];
  }
  float r  = sigf_(gi_r + gh_r);
  float z  = sigf_(gi_z + gh_z);
  float nn = tanhf(gi_n + r*gh_n);
  X[n*32 + f] = (1.f - z)*nn + z*h;
}

// g = q_star @ wi + hs @ wh + b  (G x 128)
__global__ void k_lstm_g(const float* __restrict__ qstar, const float* __restrict__ hs,
                         const float* __restrict__ wi, const float* __restrict__ wh,
                         const float* __restrict__ b, float* __restrict__ g4, int G){
  int t = blockIdx.x*blockDim.x + threadIdx.x;
  if (t >= G*128) return;
  int g = t >> 7, col = t & 127;
  float a = b[col];
#pragma unroll 4
  for (int k = 0; k < 64; ++k) a += qstar[g*64+k]*wi[k*128+col];
#pragma unroll 4
  for (int k = 0; k < 32; ++k) a += hs[g*32+k]*wh[k*128+col];
  g4[t] = a;
}

__global__ void k_lstm_gate(const float* __restrict__ g4, float* __restrict__ cs,
                            float* __restrict__ hs, int G){
  int t = blockIdx.x*blockDim.x + threadIdx.x;
  if (t >= G*32) return;
  int g = t >> 5, f = t & 31;
  float i  = g4[g*128 + f];
  float fo = g4[g*128 + 32 + f];
  float gt = g4[g*128 + 64 + f];
  float o  = g4[g*128 + 96 + f];
  float c = sigf_(fo)*cs[t] + sigf_(i)*tanhf(gt);
  cs[t] = c;
  hs[t] = sigf_(o)*tanhf(c);
}

// Segment softmax attention. batch = arange(N)*G//N is monotone -> contiguous ranges.
__global__ void k_attn(const float* __restrict__ X, const float* __restrict__ hs,
                       float* __restrict__ rbuf, int N, int G){
  int g = blockIdx.x;
  int tid = threadIdx.x;
  int s0 = (int)(((long long)g*N + G - 1) / G);
  int s1 = (int)(((long long)(g+1)*N + G - 1) / G);
  __shared__ float q[32];
  __shared__ float red[256];
  __shared__ float rsh[32];
  if (tid < 32) q[tid] = hs[g*32 + tid];
  __syncthreads();
  float mx = -3.4e38f;
  for (int n = s0 + tid; n < s1; n += 256){
    float e = 0.f;
#pragma unroll
    for (int f = 0; f < 32; ++f) e += X[n*32+f]*q[f];
    mx = fmaxf(mx, e);
  }
  red[tid] = mx; __syncthreads();
  for (int st = 128; st > 0; st >>= 1){
    if (tid < st) red[tid] = fmaxf(red[tid], red[tid+st]);
    __syncthreads();
  }
  float mxall = red[0]; __syncthreads();
  float sm = 0.f;
  for (int n = s0 + tid; n < s1; n += 256){
    float e = 0.f;
#pragma unroll
    for (int f = 0; f < 32; ++f) e += X[n*32+f]*q[f];
    sm += __expf(e - mxall);
  }
  red[tid] = sm; __syncthreads();
  for (int st = 128; st > 0; st >>= 1){
    if (tid < st) red[tid] += red[tid+st];
    __syncthreads();
  }
  float small_ = red[0]; __syncthreads();
  if (tid < 32) rsh[tid] = 0.f;
  __syncthreads();
  float racc[32];
#pragma unroll
  for (int f = 0; f < 32; ++f) racc[f] = 0.f;
  for (int n = s0 + tid; n < s1; n += 256){
    float e = 0.f;
#pragma unroll
    for (int f = 0; f < 32; ++f) e += X[n*32+f]*q[f];
    float a = __expf(e - mxall) / small_;
#pragma unroll
    for (int f = 0; f < 32; ++f) racc[f] += a * X[n*32+f];
  }
#pragma unroll
  for (int f = 0; f < 32; ++f) atomicAdd(&rsh[f], racc[f]);
  __syncthreads();
  if (tid < 32) rbuf[g*32 + tid] = rsh[tid];
}

__global__ void k_qstar(const float* __restrict__ hs, const float* __restrict__ rbuf,
                        float* __restrict__ qstar, int G){
  int t = blockIdx.x*blockDim.x + threadIdx.x;
  if (t >= G*64) return;
  int g = t >> 6, k = t & 63;
  qstar[t] = (k < 32) ? hs[g*32 + k] : rbuf[g*32 + (k - 32)];
}

__global__ void k_mem_gate(const float* __restrict__ g4, const float* __restrict__ cx,
                           float* __restrict__ out_h, float* __restrict__ out_c,
                           float* __restrict__ hbuf, int G){
  int t = blockIdx.x*blockDim.x + threadIdx.x;
  if (t >= G*32) return;
  int g = t >> 5, f = t & 31;
  float i  = g4[g*128 + f];
  float fo = g4[g*128 + 32 + f];
  float gt = g4[g*128 + 64 + f];
  float o  = g4[g*128 + 96 + f];
  float c = sigf_(fo)*cx[t] + sigf_(i)*tanhf(gt);
  float h = sigf_(o)*tanhf(c);
  out_h[t] = h; out_c[t] = c; hbuf[t] = h;
}

__global__ void k_fc1(const float* __restrict__ hbuf, const float* __restrict__ w,
                      const float* __restrict__ b, float* __restrict__ fcb, int G){
  int t = blockIdx.x*blockDim.x + threadIdx.x;
  if (t >= G*32) return;
  int g = t >> 5, f = t & 31;
  float a = b[f];
#pragma unroll 4
  for (int d = 0; d < 32; ++d) a += hbuf[g*32+d]*w[d*32+f];
  fcb[t] = fmaxf(a, 0.f);
}

__global__ void k_fc2(const float* __restrict__ fcb, const float* __restrict__ w,
                      const float* __restrict__ b, float* __restrict__ v, int G){
  int g = blockIdx.x*blockDim.x + threadIdx.x;
  if (g >= G) return;
  float a = b[0];
#pragma unroll 4
  for (int d = 0; d < 32; ++d) a += fcb[g*32+d]*w[d];
  v[g] = a;
}

extern "C" void kernel_launch(void* const* d_in, const int* in_sizes, int n_in,
                              void* d_out, int out_size, void* d_ws, size_t ws_size,
                              hipStream_t stream) {
  const float* x        = (const float*)d_in[0];
  const float* ea       = (const float*)d_in[1];
  const float* hx       = (const float*)d_in[2];
  const float* cx       = (const float*)d_in[3];
  const float* lin0_w   = (const float*)d_in[4];
  const float* lin0_b   = (const float*)d_in[5];
  const float* e_w1     = (const float*)d_in[6];
  const float* e_b1     = (const float*)d_in[7];
  const float* e_w2     = (const float*)d_in[8];
  const float* e_b2     = (const float*)d_in[9];
  const float* root_w   = (const float*)d_in[10];
  const float* conv_b   = (const float*)d_in[11];
  const float* gru_wih  = (const float*)d_in[12];
  const float* gru_whh  = (const float*)d_in[13];
  const float* gru_bih  = (const float*)d_in[14];
  const float* gru_bhh  = (const float*)d_in[15];
  const float* s2s_wi   = (const float*)d_in[16];
  const float* s2s_wh   = (const float*)d_in[17];
  const float* s2s_b    = (const float*)d_in[18];
  const float* mem_wi   = (const float*)d_in[19];
  const float* mem_wh   = (const float*)d_in[20];
  const float* mem_b    = (const float*)d_in[21];
  const float* fc1_w    = (const float*)d_in[22];
  const float* fc1_b    = (const float*)d_in[23];
  const float* fc2_w    = (const float*)d_in[24];
  const float* fc2_b    = (const float*)d_in[25];
  const int*   eidx     = (const int*)d_in[26];

  const int N = in_sizes[0] / 3;
  const int E = in_sizes[1] / 4;
  const int G = in_sizes[2] / 32;
  const int tilesM = (E + 15) / 16;
  const int* src = eidx;
  const int* dst = eidx + E;

  // ---- workspace carve-out (W_e fp16 dominates at ~246 MB) ----
  size_t off = 0;
  auto carve = [&](size_t bytes) -> void* {
    void* p = (char*)d_ws + off;
    off += (bytes + 255) & ~(size_t)255;
    return p;
  };
  float*    X     = (float*)   carve((size_t)N*32*4);
  float*    aggr  = (float*)   carve((size_t)N*32*4);
  float*    deg   = (float*)   carve((size_t)N*4);
  float*    qstar = (float*)   carve((size_t)G*64*4);
  float*    hs    = (float*)   carve((size_t)G*32*4);
  float*    cs    = (float*)   carve((size_t)G*32*4);
  float*    rbuf  = (float*)   carve((size_t)G*32*4);
  float*    g4    = (float*)   carve((size_t)G*128*4);
  float*    hbuf  = (float*)   carve((size_t)G*32*4);
  float*    fcb   = (float*)   carve((size_t)G*32*4);
  _Float16* Apack = (_Float16*)carve((size_t)tilesM*512*2);
  _Float16* Bpack = (_Float16*)carve((size_t)64*512*2);
  _Float16* We    = (_Float16*)carve((size_t)E*1024*2);

  float* out_v = (float*)d_out;           // G
  float* out_h = (float*)d_out + G;       // G*32
  float* out_c = (float*)d_out + G + G*32;

  const int B = 256;
  #define GRID(n) (((n) + B - 1) / B)

  // Node embedding + edge network (WMMA GEMM producing fp16 W_e, f-major per edge)
  k_lin0   <<<GRID(N*32), B, 0, stream>>>(x, lin0_w, lin0_b, X, N);
  k_ehidden<<<GRID(E*32), B, 0, stream>>>(ea, e_w1, e_b1, Apack, E);
  k_packB  <<<GRID(32*1024), B, 0, stream>>>(e_w2, Bpack);
  k_we_gemm<<<GRID(tilesM*32), B, 0, stream>>>(Apack, Bpack, e_b2, We, tilesM);

  // Degrees
  k_zero_f32<<<GRID(N), B, 0, stream>>>(deg, N);
  k_deg     <<<GRID(E), B, 0, stream>>>(dst, deg, E);

  // 6 rounds of message passing + GRU
  for (int it = 0; it < 6; ++it){
    k_zero_f32<<<GRID(N*32), B, 0, stream>>>(aggr, N*32);
    k_msg<<<GRID(E*32), B, 0, stream>>>(We, X, src, dst, aggr, E);
    k_gru<<<GRID(N*32), B, 0, stream>>>(X, aggr, deg, root_w, conv_b,
                                        gru_wih, gru_whh, gru_bih, gru_bhh, N);
  }

  // Set2Set
  k_zero_f32<<<GRID(G*64), B, 0, stream>>>(qstar, G*64);
  k_zero_f32<<<GRID(G*32), B, 0, stream>>>(hs, G*32);
  k_zero_f32<<<GRID(G*32), B, 0, stream>>>(cs, G*32);
  for (int it = 0; it < 6; ++it){
    k_lstm_g   <<<GRID(G*128), B, 0, stream>>>(qstar, hs, s2s_wi, s2s_wh, s2s_b, g4, G);
    k_lstm_gate<<<GRID(G*32),  B, 0, stream>>>(g4, cs, hs, G);
    k_attn     <<<G, 256, 0, stream>>>(X, hs, rbuf, N, G);
    k_qstar    <<<GRID(G*64), B, 0, stream>>>(hs, rbuf, qstar, G);
  }

  // Memory LSTM + head
  k_lstm_g  <<<GRID(G*128), B, 0, stream>>>(qstar, hx, mem_wi, mem_wh, mem_b, g4, G);
  k_mem_gate<<<GRID(G*32),  B, 0, stream>>>(g4, cx, out_h, out_c, hbuf, G);
  k_fc1     <<<GRID(G*32),  B, 0, stream>>>(hbuf, fc1_w, fc1_b, fcb, G);
  k_fc2     <<<GRID(G),     B, 0, stream>>>(fcb, fc2_w, fc2_b, out_v, G);
  #undef GRID
}
